// MagneticTopologicalTransformer_49830210568392
// MI455X (gfx1250) — compile-verified, matrix operand order checked by reading, not run on hardware
//
#include <hip/hip_runtime.h>
#include <hip/hip_bf16.h>

// ---------------------------------------------------------------------------
// MagneticTopologicalTransformer for MI455X (gfx1250, wave32, WMMA)
// ---------------------------------------------------------------------------
// N=50000 nodes, E=800000 edges, HID=128, H=4, DH=32, 3 layers, 64 graphs.
// Dense projections -> v_wmma_f32_16x16x32_bf16 (fp32 accum).
// Edge phase (memory bound) -> bf16 gathers, global fp32 atomics for scatter,
// two-pass global softmax per head with monotonic-uint atomic max.
// ---------------------------------------------------------------------------

typedef __attribute__((ext_vector_type(16))) __bf16 v16bf;
typedef __attribute__((ext_vector_type(8)))  __bf16 v8bf;
typedef __attribute__((ext_vector_type(8)))  float  v8f;

#define GT_N    50000
#define GT_E    800000
#define GT_HID  128

// ---------------- small helpers -------------------------------------------

__device__ __forceinline__ unsigned int encodeMono(float f) {
    unsigned int u = __float_as_uint(f);
    return (u & 0x80000000u) ? ~u : (u | 0x80000000u);
}
__device__ __forceinline__ float decodeMono(unsigned int m) {
    unsigned int u = (m & 0x80000000u) ? (m ^ 0x80000000u) : ~m;
    return __uint_as_float(u);
}

__global__ __launch_bounds__(256) void k_zero_f32(float* __restrict__ p, size_t n) {
    size_t i = (size_t)blockIdx.x * blockDim.x + threadIdx.x;
    if (i < n) p[i] = 0.0f;
}

__global__ __launch_bounds__(256) void k_f32_to_bf16(const float* __restrict__ s,
                                                     __bf16* __restrict__ d, size_t n) {
    size_t i = (size_t)blockIdx.x * blockDim.x + threadIdx.x;
    if (i < n) d[i] = (__bf16)s[i];
}

__global__ __launch_bounds__(256) void k_copy_f32(const float* __restrict__ s,
                                                  float* __restrict__ d, int n, int off) {
    int i = blockIdx.x * blockDim.x + threadIdx.x;
    if (i < n) d[off + i] = s[i];
}

// Pack W[K,dout] (f32, row-major) into transposed bf16 Wt[doutTotal, K]:
// Wt[(rowOff+c)*ldK + r] = W[r*dout + c]  -> B-fragment loads become contiguous.
__global__ __launch_bounds__(256) void k_pack_wt(const float* __restrict__ w,
                                                 __bf16* __restrict__ wt,
                                                 int din, int dout, int rowOff, int ldK) {
    int i = blockIdx.x * blockDim.x + threadIdx.x;
    if (i >= din * dout) return;
    int r = i / dout, c = i % dout;
    wt[(size_t)(rowOff + c) * ldK + r] = (__bf16)w[i];
}

// ---------------- WMMA GEMM ------------------------------------------------
// C[M,Nout] = act( A_bf16[M,K] @ Wt_bf16[Nout,K]^T + bias + residual )
// One wave -> one 16x16 tile.  256-thread block = 8 waves = 128 output cols.
// grid = (M/16, Nout/128).  act: 0=none 1=relu 2=gelu(exact erf)
__global__ __launch_bounds__(256) void k_gemm(const __bf16* __restrict__ A,
                                              const __bf16* __restrict__ Wt,
                                              const float*  __restrict__ bias,
                                              const float*  __restrict__ residual,
                                              float*        __restrict__ outF,
                                              __bf16*       __restrict__ outB,
                                              int M, int K, int Nout, int act) {
    const int wave = threadIdx.x >> 5;
    const int lane = threadIdx.x & 31;
    const int tm   = blockIdx.x;
    const int tn   = blockIdx.y * 8 + wave;

    const int rlo  = lane & 15;
    const int base = (lane >> 4) * 8;          // ISA 16-bit A/B fragment layout

    const __bf16* aPtr = A  + (size_t)(tm * 16 + rlo) * K;
    const __bf16* bPtr = Wt + (size_t)(tn * 16 + rlo) * K;

    v8f acc = {};
    for (int k0 = 0; k0 < K; k0 += 32) {
        v8bf alo = *(const v8bf*)(aPtr + k0 + base);
        v8bf ahi = *(const v8bf*)(aPtr + k0 + 16 + base);
        v8bf blo = *(const v8bf*)(bPtr + k0 + base);
        v8bf bhi = *(const v8bf*)(bPtr + k0 + 16 + base);
        if (k0 + 32 < K) __builtin_prefetch(aPtr + k0 + 32, 0, 0);   // global_prefetch_b8
        v16bf a, b;
#pragma unroll
        for (int i = 0; i < 8; ++i) {
            a[i] = alo[i]; a[8 + i] = ahi[i];
            b[i] = blo[i]; b[8 + i] = bhi[i];
        }
        acc = __builtin_amdgcn_wmma_f32_16x16x32_bf16(false, a, false, b,
                                                      (short)0, acc, false, false);
    }

    const int colOut = tn * 16 + rlo;
    const float bv = bias ? bias[colOut] : 0.0f;
#pragma unroll
    for (int i = 0; i < 8; ++i) {
        int rowOut = tm * 16 + i + ((lane >> 4) << 3);   // C/D layout: lanes16-31 -> M+8
        float v = acc[i] + bv;
        if (residual) v += residual[(size_t)rowOut * Nout + colOut];
        if (act == 1)      v = fmaxf(v, 0.0f);
        else if (act == 2) v = 0.5f * v * (1.0f + erff(v * 0.70710678118f));
        size_t o = (size_t)rowOut * Nout + colOut;
        if (outF) outF[o] = v;
        if (outB) outB[o] = (__bf16)v;
    }
}

// ---------------- LayerNorm (wave per row, 128 cols) -----------------------
__global__ __launch_bounds__(256) void k_layernorm(const float* __restrict__ in,
                                                   const float* __restrict__ g,
                                                   const float* __restrict__ b,
                                                   __bf16* __restrict__ out, int M) {
    const int wave = threadIdx.x >> 5, lane = threadIdx.x & 31;
    const int row = blockIdx.x * 8 + wave;
    if (row >= M) return;
    const float* p = in + (size_t)row * GT_HID;
    float v[4], s = 0.0f;
#pragma unroll
    for (int i = 0; i < 4; ++i) { v[i] = p[lane * 4 + i]; s += v[i]; }
#pragma unroll
    for (int m = 16; m >= 1; m >>= 1) s += __shfl_xor(s, m, 32);
    const float mean = s * (1.0f / 128.0f);
    float var = 0.0f;
#pragma unroll
    for (int i = 0; i < 4; ++i) { float d = v[i] - mean; var += d * d; }
#pragma unroll
    for (int m = 16; m >= 1; m >>= 1) var += __shfl_xor(var, m, 32);
    const float r = rsqrtf(var * (1.0f / 128.0f) + 1e-5f);
#pragma unroll
    for (int i = 0; i < 4; ++i) {
        int c = lane * 4 + i;
        out[(size_t)row * GT_HID + c] = (__bf16)((v[i] - mean) * r * g[c] + b[c]);
    }
}

// ---------------- input projection (K=7 and K=12, VALU) --------------------
__global__ __launch_bounds__(256) void k_inproj(const float* __restrict__ x,
                                                const float* __restrict__ sf,
                                                const float* __restrict__ Wn,
                                                const float* __restrict__ bn,
                                                const float* __restrict__ Ws,
                                                const float* __restrict__ bs,
                                                float* __restrict__ h, int Nn) {
    int i = blockIdx.x * blockDim.x + threadIdx.x;
    if (i >= Nn * GT_HID) return;
    int n = i >> 7, c = i & 127;
    float acc = bn[c] + bs[c];
    const float* xp = x + (size_t)n * 7;
#pragma unroll
    for (int j = 0; j < 7; ++j)  acc += xp[j] * Wn[j * GT_HID + c];
    const float* sp = sf + (size_t)n * 12;
#pragma unroll
    for (int j = 0; j < 12; ++j) acc += sp[j] * Ws[j * GT_HID + c];
    h[i] = acc;
}

// ---------------- edge MLP: [E,4] -> relu(32) -> [E,4] ---------------------
__global__ __launch_bounds__(256) void k_edge_mlp(const float* __restrict__ ea,
                                                  const float* __restrict__ w1,
                                                  const float* __restrict__ b1,
                                                  const float* __restrict__ w2,
                                                  const float* __restrict__ b2,
                                                  float* __restrict__ ew, int E) {
    __shared__ float sw1[4 * 32], sb1[32], sw2[32 * 4], sb2[4];
    int t = threadIdx.x;
    if (t < 128) { sw1[t] = w1[t]; sw2[t] = w2[t]; }
    if (t >= 128 && t < 160) sb1[t - 128] = b1[t - 128];
    if (t < 4) sb2[t] = b2[t];
    __syncthreads();
    int e = blockIdx.x * blockDim.x + t;
    if (e >= E) return;
    const float a0 = ea[(size_t)e * 4 + 0], a1 = ea[(size_t)e * 4 + 1];
    const float a2 = ea[(size_t)e * 4 + 2], a3 = ea[(size_t)e * 4 + 3];
    float o0 = sb2[0], o1 = sb2[1], o2 = sb2[2], o3 = sb2[3];
#pragma unroll
    for (int j = 0; j < 32; ++j) {
        float hj = a0 * sw1[j] + a1 * sw1[32 + j] + a2 * sw1[64 + j] + a3 * sw1[96 + j] + sb1[j];
        hj = fmaxf(hj, 0.0f);
        o0 += hj * sw2[j * 4 + 0]; o1 += hj * sw2[j * 4 + 1];
        o2 += hj * sw2[j * 4 + 2]; o3 += hj * sw2[j * 4 + 3];
    }
    ew[(size_t)e * 4 + 0] = o0; ew[(size_t)e * 4 + 1] = o1;
    ew[(size_t)e * 4 + 2] = o2; ew[(size_t)e * 4 + 3] = o3;
}

// ---------------- attention scores (wave per edge) + global max ------------
// attBuf holds ew on input, att on output (in place).
__global__ __launch_bounds__(256) void k_score(const __bf16* __restrict__ qkv,
                                               const int* __restrict__ ei,
                                               float* __restrict__ attBuf,
                                               unsigned int* __restrict__ headMax, int E) {
    __shared__ unsigned int smax[4];
    int t = threadIdx.x;
    if (t < 4) smax[t] = 0u;
    __syncthreads();
    const int wave = t >> 5, lane = t & 31;
    const int e = blockIdx.x * 8 + wave;
    if (e < E) {
        const int src = ei[e], dst = ei[E + e];
        const __bf16* qp = qkv + (size_t)dst * 384 + lane * 4;          // q block
        const __bf16* kp = qkv + (size_t)src * 384 + 128 + lane * 4;    // k block
        float p = 0.0f;
#pragma unroll
        for (int i = 0; i < 4; ++i) p += (float)qp[i] * (float)kp[i];
        p += __shfl_xor(p, 1, 32); p += __shfl_xor(p, 2, 32); p += __shfl_xor(p, 4, 32);
        if ((lane & 7) == 0) {
            int h = lane >> 3;
            float att = p * 0.17677669529f * attBuf[(size_t)e * 4 + h]; // 1/sqrt(32) * ew
            attBuf[(size_t)e * 4 + h] = att;
            atomicMax(&smax[h], encodeMono(att));
        }
    }
    __syncthreads();
    if (t < 4 && smax[t]) atomicMax(&headMax[t], smax[t]);
}

// ---------------- exp + global sum -----------------------------------------
__global__ __launch_bounds__(256) void k_expsum(float* __restrict__ attBuf,
                                                const unsigned int* __restrict__ headMax,
                                                float* __restrict__ headSum, int total) {
    __shared__ float ssum[4];
    int t = threadIdx.x;
    if (t < 4) ssum[t] = 0.0f;
    __syncthreads();
    int i = blockIdx.x * blockDim.x + t;
    if (i < total) {
        int h = i & 3;
        float v = __expf(attBuf[i] - decodeMono(headMax[h]));
        attBuf[i] = v;
        atomicAdd(&ssum[h], v);                // ds_add_f32
    }
    __syncthreads();
    if (t < 4) unsafeAtomicAdd(&headSum[t], ssum[t]);
}

// ---------------- message scatter (wave per edge, fp32 HW atomics) ---------
__global__ __launch_bounds__(256) void k_scatter(const __bf16* __restrict__ qkv,
                                                 const int* __restrict__ ei,
                                                 const float* __restrict__ attBuf,
                                                 const float* __restrict__ headSum,
                                                 float* __restrict__ agg, int E) {
    const int t = threadIdx.x, wave = t >> 5, lane = t & 31;
    const int e = blockIdx.x * 8 + wave;
    if (e >= E) return;
    const int src = ei[e], dst = ei[E + e];
    const int h = lane >> 3;
    const float alpha = attBuf[(size_t)e * 4 + h] / headSum[h];
    const __bf16* vp = qkv + (size_t)src * 384 + 256 + lane * 4;        // v block
    float* ap = agg + (size_t)dst * GT_HID + lane * 4;
#pragma unroll
    for (int i = 0; i < 4; ++i) unsafeAtomicAdd(&ap[i], alpha * (float)vp[i]);
}

// ---------------- graph pooling --------------------------------------------
__global__ __launch_bounds__(256) void k_pool(const float* __restrict__ h,
                                              const int* __restrict__ batch,
                                              float* __restrict__ pooled,
                                              float* __restrict__ cnt, int Nn) {
    int i = blockIdx.x * blockDim.x + threadIdx.x;
    if (i >= Nn * GT_HID) return;
    int n = i >> 7, c = i & 127;
    int b = batch[n];
    unsafeAtomicAdd(&pooled[(size_t)b * GT_HID + c], h[i]);
    if (c == 0) unsafeAtomicAdd(&cnt[b], 1.0f);
}

// ---------------- prediction heads (per-graph block) -----------------------
__global__ __launch_bounds__(128) void k_heads(const float* __restrict__ pooled,
                                               const float* __restrict__ cnt,
                                               const float* __restrict__ m1w, const float* __restrict__ m1b,
                                               const float* __restrict__ m2w, const float* __restrict__ m2b,
                                               const float* __restrict__ t1w, const float* __restrict__ t1b,
                                               const float* __restrict__ t2w, const float* __restrict__ t2b,
                                               float* __restrict__ out) {
    __shared__ float hg[128], hid[64];
    const int g = blockIdx.x, t = threadIdx.x;
    const float c = fmaxf(cnt[g], 1.0f);
    hg[t] = pooled[(size_t)g * GT_HID + t] / c;
    __syncthreads();
    if (t < 64) {
        float a = m1b[t];
        for (int j = 0; j < 128; ++j) a += hg[j] * m1w[j * 64 + t];
        hid[t] = fmaxf(a, 0.0f);
    }
    __syncthreads();
    if (t < 3) {
        float a = m2b[t];
        for (int j = 0; j < 64; ++j) a += hid[j] * m2w[j * 3 + t];
        out[g * 3 + t] = a;                        // mag: first 64*3 floats
    }
    __syncthreads();
    if (t < 64) {
        float a = t1b[t];
        for (int j = 0; j < 128; ++j) a += hg[j] * t1w[j * 64 + t];
        hid[t] = fmaxf(a, 0.0f);
    }
    __syncthreads();
    if (t < 2) {
        float a = t2b[t];
        for (int j = 0; j < 64; ++j) a += hid[j] * t2w[j * 2 + t];
        out[192 + g * 2 + t] = a;                  // top: next 64*2 floats
    }
}

// ---------------------------------------------------------------------------
// host driver
// ---------------------------------------------------------------------------
extern "C" void kernel_launch(void* const* d_in, const int* in_sizes, int n_in,
                              void* d_out, int out_size, void* d_ws, size_t ws_size,
                              hipStream_t stream) {
    (void)in_sizes; (void)n_in; (void)out_size; (void)ws_size;
    const int N = GT_N, E = GT_E;

    auto F = [&](int i) -> const float* { return (const float*)d_in[i]; };
    // setup_inputs() insertion order, params flattened depth-first in insertion order
    const float* x   = F(0);
    const float* ea  = F(1);
    const float* sf  = F(2);
    const float* Wnp = F(3);  const float* bnp = F(4);
    const float* Wsp = F(5);  const float* bsp = F(6);
    auto LP = [&](int l, int j) -> const float* { return F(7 + l * 20 + j); };
    const float* m1w = F(67); const float* m1b = F(68);
    const float* m2w = F(69); const float* m2b = F(70);
    const float* t1w = F(71); const float* t1b = F(72);
    const float* t2w = F(73); const float* t2b = F(74);
    const int* ei    = (const int*)d_in[75];
    const int* batch = (const int*)d_in[76];

    // bump allocator over workspace
    char* p = (char*)d_ws;
    auto alloc = [&](size_t bytes) -> char* {
        char* r = p; p += (bytes + 255) & ~(size_t)255; return r;
    };
    float*  hB      = (float*)alloc((size_t)N * 128 * 4);
    float*  outRes  = (float*)alloc((size_t)N * 128 * 4);
    float*  aggF    = (float*)alloc((size_t)N * 128 * 4);
    __bf16* hnB     = (__bf16*)alloc((size_t)N * 128 * 2);
    __bf16* aggBf   = (__bf16*)alloc((size_t)N * 128 * 2);
    __bf16* ffnB    = (__bf16*)alloc((size_t)N * 256 * 2);
    __bf16* qkvB    = (__bf16*)alloc((size_t)N * 384 * 2);
    float*  attB    = (float*)alloc((size_t)E * 4 * 4);
    __bf16* wtB     = (__bf16*)alloc((size_t)384 * 256 * 2);
    float*  biasB   = (float*)alloc(384 * 4);
    unsigned int* headMax = (unsigned int*)alloc(4 * 4);
    float*  headSum = (float*)alloc(4 * 4);
    float*  pooled  = (float*)alloc(64 * 128 * 4);
    float*  cnt     = (float*)alloc(64 * 4);

    // h0 = x @ Wnode + structure @ Wstruct
    k_inproj<<<(N * 128 + 255) / 256, 256, 0, stream>>>(x, sf, Wnp, bnp, Wsp, bsp, hB, N);

    for (int l = 0; l < 3; ++l) {
        const float *qw = LP(l,0), *qb = LP(l,1), *kw = LP(l,2), *kb = LP(l,3);
        const float *vw = LP(l,4), *vb = LP(l,5);
        const float *e1w = LP(l,6), *e1b = LP(l,7), *e2w = LP(l,8), *e2b = LP(l,9);
        const float *ow = LP(l,10), *ob = LP(l,11);
        const float *g1 = LP(l,12), *bt1 = LP(l,13), *g2 = LP(l,14), *bt2 = LP(l,15);
        const float *f1w = LP(l,16), *f1b = LP(l,17), *f2w = LP(l,18), *f2b = LP(l,19);

        // LN1 -> bf16
        k_layernorm<<<N / 8, 256, 0, stream>>>(hB, g1, bt1, hnB, N);

        // fused q|k|v GEMM: pack Wt[384,128] + bias[384]
        const int nb = (128 * 128 + 255) / 256;
        k_pack_wt<<<nb, 256, 0, stream>>>(qw, wtB, 128, 128,   0, 128);
        k_pack_wt<<<nb, 256, 0, stream>>>(kw, wtB, 128, 128, 128, 128);
        k_pack_wt<<<nb, 256, 0, stream>>>(vw, wtB, 128, 128, 256, 128);
        k_copy_f32<<<1, 128, 0, stream>>>(qb, biasB, 128, 0);
        k_copy_f32<<<1, 128, 0, stream>>>(kb, biasB, 128, 128);
        k_copy_f32<<<1, 128, 0, stream>>>(vb, biasB, 128, 256);
        k_gemm<<<dim3(N / 16, 3), 256, 0, stream>>>(hnB, wtB, biasB, nullptr,
                                                    nullptr, qkvB, N, 128, 384, 0);

        // edge weights ew -> attB
        k_edge_mlp<<<(E + 255) / 256, 256, 0, stream>>>(ea, e1w, e1b, e2w, e2b, attB, E);

        // global softmax over all edges per head (two passes)
        k_zero_f32<<<1, 32, 0, stream>>>((float*)headMax, 4);
        k_zero_f32<<<1, 32, 0, stream>>>(headSum, 4);
        k_score<<<E / 8, 256, 0, stream>>>(qkvB, ei, attB, headMax, E);
        k_expsum<<<(E * 4 + 255) / 256, 256, 0, stream>>>(attB, headMax, headSum, E * 4);

        // scatter messages
        k_zero_f32<<<(N * 128 + 255) / 256, 256, 0, stream>>>(aggF, (size_t)N * 128);
        k_scatter<<<E / 8, 256, 0, stream>>>(qkvB, ei, attB, headSum, aggF, E);
        k_f32_to_bf16<<<(N * 128 + 255) / 256, 256, 0, stream>>>(aggF, aggBf, (size_t)N * 128);

        // out projection + residual h
        k_pack_wt<<<nb, 256, 0, stream>>>(ow, wtB, 128, 128, 0, 128);
        k_gemm<<<dim3(N / 16, 1), 256, 0, stream>>>(aggBf, wtB, ob, hB,
                                                    outRes, nullptr, N, 128, 128, 0);

        // LN2 + FFN (gelu) + residual
        k_layernorm<<<N / 8, 256, 0, stream>>>(outRes, g2, bt2, hnB, N);
        k_pack_wt<<<(128 * 256 + 255) / 256, 256, 0, stream>>>(f1w, wtB, 128, 256, 0, 128);
        k_gemm<<<dim3(N / 16, 2), 256, 0, stream>>>(hnB, wtB, f1b, nullptr,
                                                    nullptr, ffnB, N, 128, 256, 2);
        k_pack_wt<<<(256 * 128 + 255) / 256, 256, 0, stream>>>(f2w, wtB, 256, 128, 0, 256);
        k_gemm<<<dim3(N / 16, 1), 256, 0, stream>>>(ffnB, wtB, f2b, outRes,
                                                    hB, nullptr, N, 256, 128, 0);
    }

    // scatter-mean pooling + heads
    k_zero_f32<<<(64 * 128 + 255) / 256, 256, 0, stream>>>(pooled, 64 * 128);
    k_zero_f32<<<1, 64, 0, stream>>>(cnt, 64);
    k_pool<<<(N * 128 + 255) / 256, 256, 0, stream>>>(hB, batch, pooled, cnt, N);
    k_heads<<<64, 128, 0, stream>>>(pooled, cnt, m1w, m1b, m2w, m2b,
                                    t1w, t1b, t2w, t2b, (float*)d_out);
}
